// TwinBottleneck_8443905704331
// MI455X (gfx1250) — matrix operand order repaired and checked by reading, not compiled, not run on hardware
//
#include <hip/hip_runtime.h>
#include <hip/hip_bf16.h>
#include <math.h>

// ---------------------------------------------------------------------------
// TwinBottleneck fused pipeline for MI455X (gfx1250, wave32, WMMA bf16)
//   out = sigmoid( D^-1/2 A D^-1/2 @ (cbn@W.T + b) ),  A = (1-|2G-s_i-s_j|/64)^1.4
// Dominant GEMM (8192x8192x512, 68.7 GFLOP) runs on v_wmma_f32_16x16x32_bf16.
// adj materialized as bf16 (128 MB — fits the 192 MB L2).
// ---------------------------------------------------------------------------

typedef __bf16 bhalf;
typedef __attribute__((ext_vector_type(16))) __bf16 v16bf;
typedef __attribute__((ext_vector_type(8)))  __bf16 v8bf;
typedef __attribute__((ext_vector_type(8)))  float  v8f;

#define CAT16(lo, hi) __builtin_shufflevector((lo), (hi), 0,1,2,3,4,5,6,7,8,9,10,11,12,13,14,15)
#define WMMA_BF16(a, b, c) \
  __builtin_amdgcn_wmma_f32_16x16x32_bf16(false, (a), false, (b), (short)0, (c), false, false)

#define NROW 8192
#define DDIM 64
#define CDIM 512

// ---------------------------------------------------------------------------
// K1: s = rowsum(bbn); bf16 copies of bbn, cbn, W.
// ---------------------------------------------------------------------------
__global__ void k_prep(const float* __restrict__ bbn, const float* __restrict__ cbn,
                       const float* __restrict__ W, bhalf* __restrict__ bbnh,
                       bhalf* __restrict__ cbh, bhalf* __restrict__ Wh,
                       float* __restrict__ s) {
  int idx = blockIdx.x * blockDim.x + threadIdx.x;
  int stride = gridDim.x * blockDim.x;
  for (int i = idx; i < NROW * CDIM; i += stride) cbh[i] = (bhalf)cbn[i];
  for (int i = idx; i < CDIM * CDIM; i += stride) Wh[i] = (bhalf)W[i];
  for (int i = idx; i < NROW * DDIM; i += stride) bbnh[i] = (bhalf)bbn[i];
  for (int i = idx; i < NROW; i += stride) {
    float acc = 0.f;
#pragma unroll
    for (int k = 0; k < DDIM; ++k) acc += bbn[i * DDIM + k];
    s[i] = acc;
  }
}

// ---------------------------------------------------------------------------
// K2: fc = cbn @ W.T + b.   One 16x16 tile per wave, K=512 (16 WMMAs).
// B = W.T, so B-columns are W-rows: contiguous per-lane b128 loads.
// ---------------------------------------------------------------------------
__global__ void k_fc(const bhalf* __restrict__ cbh, const bhalf* __restrict__ Wh,
                     const float* __restrict__ bias, float* __restrict__ fc) {
  int wid  = (blockIdx.x * blockDim.x + threadIdx.x) >> 5;   // 0..16383
  int lane = threadIdx.x & 31;
  int half = lane >> 4, m = lane & 15;
  int tm = wid & 511;   // 512 M-tiles
  int tn = wid >> 9;    // 32  N-tiles

  const bhalf* arow = cbh + (size_t)(tm * 16 + m) * CDIM;  // A row (M = m)
  const bhalf* brow = Wh  + (size_t)(tn * 16 + m) * CDIM;  // B col n = m -> W row

  v8f acc = {0.f, 0.f, 0.f, 0.f, 0.f, 0.f, 0.f, 0.f};
#pragma unroll 4
  for (int kk = 0; kk < CDIM / 32; ++kk) {
    v8bf alo = *(const v8bf*)(arow + kk * 32 + half * 16);
    v8bf ahi = *(const v8bf*)(arow + kk * 32 + half * 16 + 8);
    v8bf blo = *(const v8bf*)(brow + kk * 32 + half * 16);
    v8bf bhi = *(const v8bf*)(brow + kk * 32 + half * 16 + 8);
    v16bf a = CAT16(alo, ahi);
    v16bf b = CAT16(blo, bhi);
    acc = WMMA_BF16(a, b, acc);
  }
  float bv = bias[tn * 16 + m];
  float* orow = fc + (size_t)(tm * 16 + 8 * half) * CDIM + tn * 16 + m;
#pragma unroll
  for (int v = 0; v < 8; ++v) orow[(size_t)v * CDIM] = acc[v] + bv;
}

// ---------------------------------------------------------------------------
// K3: adj tiles via G = bbn@bbn.T (2 chained WMMAs, K=64), elementwise
//     (1-|2G-s_i-s_j|/64)^1.4, store adj bf16 row-major; per-wave rowsum
//     partials (deterministic — no float atomics).
// One block per 16-row strip; wave w sweeps j in [w*1024, w*1024+1024).
// ---------------------------------------------------------------------------
__global__ void k_adj(const bhalf* __restrict__ bbnh, const float* __restrict__ s,
                      bhalf* __restrict__ adjw, float* __restrict__ dpart) {
  int ib   = blockIdx.x;          // 512 row-strips
  int w    = threadIdx.x >> 5;    // 8 waves
  int lane = threadIdx.x & 31;
  int half = lane >> 4, m = lane & 15;

  const bhalf* arow = bbnh + (size_t)(ib * 16 + m) * DDIM;
  v16bf A0 = CAT16(*(const v8bf*)(arow + half * 16),      *(const v8bf*)(arow + half * 16 + 8));
  v16bf A1 = CAT16(*(const v8bf*)(arow + 32 + half * 16), *(const v8bf*)(arow + 32 + half * 16 + 8));

  float si[8];
#pragma unroll
  for (int v = 0; v < 8; ++v) si[v] = s[ib * 16 + v + 8 * half];
  float rs[8] = {0.f, 0.f, 0.f, 0.f, 0.f, 0.f, 0.f, 0.f};
  const v8f zero = {0.f, 0.f, 0.f, 0.f, 0.f, 0.f, 0.f, 0.f};

  for (int jt = w * 64; jt < w * 64 + 64; ++jt) {
    int j0 = jt * 16;
    const bhalf* brow = bbnh + (size_t)(j0 + m) * DDIM;  // B col n = m -> bbn row j0+m
    v16bf B0 = CAT16(*(const v8bf*)(brow + half * 16),      *(const v8bf*)(brow + half * 16 + 8));
    v16bf B1 = CAT16(*(const v8bf*)(brow + 32 + half * 16), *(const v8bf*)(brow + 32 + half * 16 + 8));
    v8f g = WMMA_BF16(A0, B0, zero);
    g = WMMA_BF16(A1, B1, g);

    float sj = s[j0 + m];
    bhalf* acol = adjw + (size_t)(ib * 16 + 8 * half) * NROW + j0 + m;
#pragma unroll
    for (int v = 0; v < 8; ++v) {
      float x = 2.f * g[v] - si[v] - sj;
      float t = 1.f - fabsf(x) * (1.f / 64.f);
      float a = exp2f(1.4f * log2f(t));       // matches pow semantics incl. t<=0
      rs[v] += a;
      acol[(size_t)v * NROW] = (bhalf)a;      // C-layout -> row-major bf16 scatter
    }
  }
  // reduce rowsums across the 16 lanes of each half (column dimension)
#pragma unroll
  for (int off = 8; off >= 1; off >>= 1)
#pragma unroll
    for (int v = 0; v < 8; ++v) rs[v] += __shfl_xor(rs[v], off, 32);
  if (m == 0)
#pragma unroll
    for (int v = 0; v < 8; ++v) dpart[(size_t)(ib * 16 + v + 8 * half) * 8 + w] = rs[v];
}

// ---------------------------------------------------------------------------
// K4: dis = rsqrt(sum(dpart)+1e-8); gT[n][j] = bf16(dis_j * fc[j][n])
//     via 32x33 LDS transpose so K5 B-fragments are contiguous.
// ---------------------------------------------------------------------------
__global__ void k_scale_t(const float* __restrict__ fc, const float* __restrict__ dpart,
                          float* __restrict__ disv, bhalf* __restrict__ gT) {
  __shared__ float tile[32][33];
  int jb = blockIdx.x;            // 256 j-blocks
  int nb = blockIdx.y;            // 16  n-blocks
  int tx = threadIdx.x & 31;
  int ty = threadIdx.x >> 5;      // 0..7
#pragma unroll
  for (int r = 0; r < 4; ++r) {
    int j = jb * 32 + ty + r * 8;
    float dsum = 0.f;
#pragma unroll
    for (int w = 0; w < 8; ++w) dsum += dpart[(size_t)j * 8 + w];
    float dj = rsqrtf(dsum + 1e-8f);
    if (nb == 0 && tx == 0) disv[j] = dj;
    tile[ty + r * 8][tx] = fc[(size_t)j * CDIM + nb * 32 + tx] * dj;
  }
  __syncthreads();
#pragma unroll
  for (int r = 0; r < 4; ++r) {
    int n = nb * 32 + ty + r * 8;
    int j = jb * 32 + tx;
    gT[(size_t)n * NROW + j] = (bhalf)tile[tx][ty + r * 8];
  }
}

// ---------------------------------------------------------------------------
// K5: out = sigmoid( dis_i * (adj @ gT.T) ).  Block = one 16-row strip x all
// 512 columns; wave w owns a 64-column group (4 accum tiles). The adj
// A-fragment is identical for all 8 waves -> L2/WGP$ hot.
// ---------------------------------------------------------------------------
__global__ void k_conv(const bhalf* __restrict__ adjw, const bhalf* __restrict__ gT,
                       const float* __restrict__ disv, float* __restrict__ out) {
  int ib   = blockIdx.x;          // 512 row-strips
  int w    = threadIdx.x >> 5;    // 8 waves -> 64-col group
  int lane = threadIdx.x & 31;
  int half = lane >> 4, m = lane & 15;

  const bhalf* arow = adjw + (size_t)(ib * 16 + m) * NROW;
  const bhalf* brow0 = gT + (size_t)(w * 64 +  0 + m) * NROW;
  const bhalf* brow1 = gT + (size_t)(w * 64 + 16 + m) * NROW;
  const bhalf* brow2 = gT + (size_t)(w * 64 + 32 + m) * NROW;
  const bhalf* brow3 = gT + (size_t)(w * 64 + 48 + m) * NROW;

  const v8f zero = {0.f, 0.f, 0.f, 0.f, 0.f, 0.f, 0.f, 0.f};
  v8f acc0 = zero, acc1 = zero, acc2 = zero, acc3 = zero;

  for (int kk = 0; kk < NROW / 32; ++kk) {
    int j0 = kk * 32;
    __builtin_prefetch(arow + j0 + 2048, 0, 1);   // global_prefetch ahead on adj row
    v16bf A = CAT16(*(const v8bf*)(arow + j0 + half * 16),
                    *(const v8bf*)(arow + j0 + half * 16 + 8));
    v16bf B0 = CAT16(*(const v8bf*)(brow0 + j0 + half * 16),
                     *(const v8bf*)(brow0 + j0 + half * 16 + 8));
    acc0 = WMMA_BF16(A, B0, acc0);
    v16bf B1 = CAT16(*(const v8bf*)(brow1 + j0 + half * 16),
                     *(const v8bf*)(brow1 + j0 + half * 16 + 8));
    acc1 = WMMA_BF16(A, B1, acc1);
    v16bf B2 = CAT16(*(const v8bf*)(brow2 + j0 + half * 16),
                     *(const v8bf*)(brow2 + j0 + half * 16 + 8));
    acc2 = WMMA_BF16(A, B2, acc2);
    v16bf B3 = CAT16(*(const v8bf*)(brow3 + j0 + half * 16),
                     *(const v8bf*)(brow3 + j0 + half * 16 + 8));
    acc3 = WMMA_BF16(A, B3, acc3);
  }

  float di[8];
#pragma unroll
  for (int v = 0; v < 8; ++v) di[v] = disv[ib * 16 + v + 8 * half];
  float* obase = out + (size_t)(ib * 16 + 8 * half) * CDIM + w * 64 + m;
#pragma unroll
  for (int v = 0; v < 8; ++v) {
    float x0 = di[v] * acc0[v];
    float x1 = di[v] * acc1[v];
    float x2 = di[v] * acc2[v];
    float x3 = di[v] * acc3[v];
    float* o = obase + (size_t)v * CDIM;
    o[0]  = 1.f / (1.f + expf(-x0));
    o[16] = 1.f / (1.f + expf(-x1));
    o[32] = 1.f / (1.f + expf(-x2));
    o[48] = 1.f / (1.f + expf(-x3));
  }
}

// ---------------------------------------------------------------------------
extern "C" void kernel_launch(void* const* d_in, const int* in_sizes, int n_in,
                              void* d_out, int out_size, void* d_ws, size_t ws_size,
                              hipStream_t stream) {
  const float* bbn  = (const float*)d_in[0];   // [8192, 64]
  const float* cbn  = (const float*)d_in[1];   // [8192, 512]
  const float* W    = (const float*)d_in[2];   // [512, 512]
  const float* bias = (const float*)d_in[3];   // [512]
  float* out = (float*)d_out;                  // [8192, 512] fp32

  char* p = (char*)d_ws;
  bhalf* bbnh = (bhalf*)p;  p += (size_t)NROW * DDIM * 2;            //   1 MB
  bhalf* cbh  = (bhalf*)p;  p += (size_t)NROW * CDIM * 2;            //   8 MB
  bhalf* Wh   = (bhalf*)p;  p += (size_t)CDIM * CDIM * 2;            // 0.5 MB
  float* s    = (float*)p;  p += (size_t)NROW * 4;                   //  32 KB
  float* dprt = (float*)p;  p += (size_t)NROW * 8 * 4;               // 256 KB
  float* disv = (float*)p;  p += (size_t)NROW * 4;                   //  32 KB
  float* fc   = (float*)p;  p += (size_t)NROW * CDIM * 4;            //  16 MB
  bhalf* gT   = (bhalf*)p;  p += (size_t)CDIM * NROW * 2;            //   8 MB
  bhalf* adjw = (bhalf*)p;                                           // 128 MB (L2-resident)

  k_prep<<<2048, 256, 0, stream>>>(bbn, cbn, W, bbnh, cbh, Wh, s);
  k_fc<<<2048, 256, 0, stream>>>(cbh, Wh, bias, fc);                 // 16384 tiles
  k_adj<<<512, 256, 0, stream>>>(bbnh, s, adjw, dprt);
  k_scale_t<<<dim3(256, 16), 256, 0, stream>>>(fc, dprt, disv, gT);
  k_conv<<<512, 256, 0, stream>>>(adjw, gT, disv, out);
}
